// ModelBlock2_16578573763178
// MI455X (gfx1250) — matrix operand order repaired and checked by reading, not compiled, not run on hardware
//
#include <hip/hip_runtime.h>
#include <hip/hip_bf16.h>

// ---------------------------------------------------------------------------
// LRU block for MI455X (gfx1250, wave32, WMMA 16x16x32 f16 -> f32 acc)
//
//   kcvt  : f32 -> f16 conversion of x, Bm, C, W (GEMM operands)
//   k1    : Bu = gamma * (x @ Bm^T)     [TDM tile->LDS, ds_load frags, WMMA]
//   k2a/b/c: 3-phase chunked diagonal scan, emits SHIFTED state hs (f16)
//   k3    : y = hs @ C^T + D*x ; z = y*rsqrt(1+y^2) -> f16   [LDS + WMMA]
//   k4    : out = z @ W^T + b + x                            [LDS + WMMA]
//
// Working set (~118MB) is L2-resident (192MB); HBM-mandatory traffic is only
// x in + out = 67MB.
// ---------------------------------------------------------------------------

typedef __attribute__((ext_vector_type(16))) _Float16 v16h;
typedef __attribute__((ext_vector_type(8)))  _Float16 v8h;
typedef __attribute__((ext_vector_type(8)))  float    v8f;
typedef __attribute__((ext_vector_type(4)))  unsigned int v4u;
typedef __attribute__((ext_vector_type(8)))  int      v8i;
typedef __attribute__((ext_vector_type(4)))  int      v4i;

#define BATCH 8
#define SEQL  8192
#define HDIM  128
#define NDIM  256
#define BL    (BATCH * SEQL)     // 65536
#define CHLEN 256
#define NCH   (SEQL / CHLEN)     // 32

// ----- fragment loaders (ISA 7.12.2 layouts, row-major K-contiguous src) ----

// A 16x32 f16: lane = m + 16*khalf; VGPR0..3 = K[kb..kb+7], VGPR4..7 = K[16+kb..]
__device__ __forceinline__ v16h load_a_f16(const _Float16* src, int ld,
                                           int row0, int k0, int lane) {
  int m  = lane & 15;
  int kb = (lane >> 4) << 3;              // 0 or 8
  const _Float16* p = src + (size_t)(row0 + m) * ld + k0;
  v8h lo = *(const v8h*)(p + kb);
  v8h hi = *(const v8h*)(p + 16 + kb);
  return __builtin_shufflevector(lo, hi, 0,1,2,3,4,5,6,7,8,9,10,11,12,13,14,15);
}

// B 32x16 f16 from weight Wt[out][K]: lane = n + 16*khalf; K[kb..kb+7], K[16+kb..]
__device__ __forceinline__ v16h load_b_f16(const _Float16* wt, int ld,
                                           int n0, int k0, int lane) {
  int n  = n0 + (lane & 15);
  int kb = k0 + ((lane >> 4) << 3);       // 0 or 8
  const _Float16* p = wt + (size_t)n * ld + kb;
  v8h lo = *(const v8h*)(p);
  v8h hi = *(const v8h*)(p + 16);
  return __builtin_shufflevector(lo, hi, 0,1,2,3,4,5,6,7,8,9,10,11,12,13,14,15);
}

#define WMMA(A, B, C) __builtin_amdgcn_wmma_f32_16x16x32_f16( \
    false, (A), false, (B), (short)0, (C), false, false)

// ----- f32 -> f16 elementwise convert ---------------------------------------
__global__ __launch_bounds__(256) void kcvt(const float* __restrict__ src,
                                            _Float16* __restrict__ dst, int n) {
  int i = blockIdx.x * 256 + threadIdx.x;
  if (i < n) dst[i] = (_Float16)src[i];
}

// ----- K1: Bu[bl][n] = gamma[n] * sum_h x[bl][h] * Bm[n][h] -----------------
// Block: 64 M-rows x 256 N.  8 waves = 4 col-groups x 2 M-strips(32).
// A tile (64x128 f16 = 16KB) staged in LDS by the Tensor Data Mover.
__global__ __launch_bounds__(256) void k1_gemm_bu(
    const _Float16* __restrict__ xh, const _Float16* __restrict__ Bmh,
    const float* __restrict__ gamma_log, float* __restrict__ Bu) {
  __shared__ _Float16 xs[64 * HDIM];      // 16 KB
  int tid  = threadIdx.x;
  int wave = tid >> 5, lane = tid & 31;
  int row0 = blockIdx.x * 64;

#if __has_builtin(__builtin_amdgcn_tensor_load_to_lds)
  if (wave == 0) {
    // D# group0: count=1 | lds_addr | global_addr[56:0] | type=2
    unsigned la = (unsigned)(size_t)xs;   // LDS aperture: low 32 bits = offset
    unsigned long long ga =
        (unsigned long long)(size_t)(xh + (size_t)row0 * HDIM);
    v4u g0 = {1u, la, (unsigned)ga,
              ((unsigned)(ga >> 32) & 0x01FFFFFFu) | 0x80000000u};
    // D# group1: data_size=2B; tensor_dim0=128, tensor_dim1=BL(65536);
    //            tile_dim0=128, tile_dim1=64; tensor_dim0_stride=128
    v8i g1 = {0x00010000,                 // wg_mask=0, data_size=1 (2 bytes)
              (int)(HDIM << 16),          // tensor_dim0[15:0]
              0,                          // tdim1[15:0]<<16 | tdim0[31:16]
              (int)((HDIM << 16) | 1),    // tile_dim0 | tensor_dim1[31:16]
              64,                         // tile_dim1
              HDIM,                       // tensor_dim0_stride[31:0]
              0, 0};
    v4i gz = {0, 0, 0, 0};
#if defined(__clang_major__) && (__clang_major__ >= 23)
    v8i gz8 = {0, 0, 0, 0, 0, 0, 0, 0};
    __builtin_amdgcn_tensor_load_to_lds(g0, g1, gz, gz, gz8, 0);
#else
    __builtin_amdgcn_tensor_load_to_lds(g0, g1, gz, gz, 0);
#endif
    __builtin_amdgcn_s_wait_tensorcnt(0);
  }
#else
  #pragma unroll
  for (int j = 0; j < 2; ++j) {           // 512 chunks of 16 halfs
    int chunk = tid + j * 256;
    int r = chunk >> 3, c = (chunk & 7) * 16;
    *(v16h*)(xs + r * HDIM + c) =
        *(const v16h*)(xh + (size_t)(row0 + r) * HDIM + c);
  }
#endif
  __syncthreads();

  int cg  = wave & 3;                     // 4 groups of 64 N columns
  int msr = (wave >> 2) * 32;             // 0 or 32 within tile
  v8f acc[2][4] = {};
  #pragma unroll
  for (int k0 = 0; k0 < HDIM; k0 += 32) {
    v16h a0 = load_a_f16(xs, HDIM, msr,      k0, lane);
    v16h a1 = load_a_f16(xs, HDIM, msr + 16, k0, lane);
    #pragma unroll
    for (int t = 0; t < 4; ++t) {
      v16h b = load_b_f16(Bmh, HDIM, cg * 64 + t * 16, k0, lane);
      acc[0][t] = WMMA(a0, b, acc[0][t]);
      acc[1][t] = WMMA(a1, b, acc[1][t]);
    }
  }
  #pragma unroll
  for (int s = 0; s < 2; ++s) {
    int mbase = row0 + msr + s * 16 + 8 * (lane >> 4);
    #pragma unroll
    for (int t = 0; t < 4; ++t) {
      int n = cg * 64 + t * 16 + (lane & 15);
      float g = __expf(gamma_log[n]);
      #pragma unroll
      for (int r = 0; r < 8; ++r)
        Bu[(size_t)(mbase + r) * NDIM + n] = acc[s][t][r] * g;
    }
  }
}

// ----- K2a: chunk-local scan ends ------------------------------------------
__global__ __launch_bounds__(256) void k2a_ends(
    const float* __restrict__ Bu, const float* __restrict__ nu_log,
    float* __restrict__ ends) {
  int wave = (blockIdx.x * 256 + threadIdx.x) >> 5;
  int lane = threadIdx.x & 31;
  int ng = wave & 7, ch = (wave >> 3) & 31, b = wave >> 8;
  int n = ng * 32 + lane;
  float A = __expf(-__expf(nu_log[n]));
  const float* p = Bu + ((size_t)b * SEQL + (size_t)ch * CHLEN) * NDIM + n;
  float h = 0.f;
  for (int i = 0; i < CHLEN; ++i) {
    __builtin_prefetch(p + (size_t)((i + 64 < CHLEN) ? i + 64 : i) * NDIM, 0, 1);
    h = fmaf(A, h, p[(size_t)i * NDIM]);
  }
  ends[((size_t)b * NCH + ch) * NDIM + n] = h;
}

// ----- K2b: scan the 32 chunk carries (carry[c] = h at last idx of chunk c-1)
__global__ __launch_bounds__(256) void k2b_carry(
    const float* __restrict__ ends, const float* __restrict__ nu_log,
    float* __restrict__ carry) {
  int wave = (blockIdx.x * 256 + threadIdx.x) >> 5;
  int lane = threadIdx.x & 31;
  int ng = wave & 7, b = wave >> 3;       // 64 waves total
  int n = ng * 32 + lane;
  float AL = __expf(-__expf(nu_log[n]) * (float)CHLEN);   // A^CHLEN
  float c = 0.f;
  for (int ch = 0; ch < NCH; ++ch) {
    size_t idx = ((size_t)b * NCH + ch) * NDIM + n;
    carry[idx] = c;
    c = fmaf(AL, c, ends[idx]);
  }
}

// ----- K2c: full scan with carry, emit shifted state hs (f16) ---------------
__global__ __launch_bounds__(256) void k2c_apply(
    const float* __restrict__ Bu, const float* __restrict__ nu_log,
    const float* __restrict__ carry, _Float16* __restrict__ hs) {
  int wave = (blockIdx.x * 256 + threadIdx.x) >> 5;
  int lane = threadIdx.x & 31;
  int ng = wave & 7, ch = (wave >> 3) & 31, b = wave >> 8;
  int n = ng * 32 + lane;
  float A = __expf(-__expf(nu_log[n]));
  size_t base = ((size_t)b * SEQL + (size_t)ch * CHLEN) * NDIM + n;
  const float* p = Bu + base;
  _Float16* q = hs + base;
  float cur = carry[((size_t)b * NCH + ch) * NDIM + n];
  for (int i = 0; i < CHLEN; ++i) {
    __builtin_prefetch(p + (size_t)((i + 64 < CHLEN) ? i + 64 : i) * NDIM, 0, 1);
    q[(size_t)i * NDIM] = (_Float16)cur;            // hs[l] = h[l-1]
    cur = fmaf(A, cur, p[(size_t)i * NDIM]);
  }
}

// ----- K3: y = hs @ C^T + D*x ; z = y*rsqrt(1+y^2) -> f16 -------------------
// Block: 128 M-rows x 128 H.  hs tile (128x256 f16 = 64KB) staged in LDS.
__global__ __launch_bounds__(256) void k3_gemm_y(
    const _Float16* __restrict__ hs, const _Float16* __restrict__ Ch,
    const float* __restrict__ D, const float* __restrict__ x,
    _Float16* __restrict__ zb) {
  __shared__ _Float16 hts[128 * NDIM];    // 64 KB
  int tid  = threadIdx.x;
  int wave = tid >> 5, lane = tid & 31;
  int row0 = blockIdx.x * 128;
  #pragma unroll
  for (int j = 0; j < 8; ++j) {           // 2048 chunks of 16 halfs
    int chunk = tid + j * 256;
    int r = chunk >> 4, c = (chunk & 15) * 16;
    *(v16h*)(hts + r * NDIM + c) =
        *(const v16h*)(hs + (size_t)(row0 + r) * NDIM + c);
  }
  __syncthreads();

  int cg  = wave & 1;                     // 2 groups of 64 H columns
  int msr = (wave >> 1) * 32;             // 0,32,64,96
  v8f acc[2][4] = {};
  #pragma unroll
  for (int k0 = 0; k0 < NDIM; k0 += 32) {
    v16h a0 = load_a_f16(hts, NDIM, msr,      k0, lane);
    v16h a1 = load_a_f16(hts, NDIM, msr + 16, k0, lane);
    #pragma unroll
    for (int t = 0; t < 4; ++t) {
      v16h b = load_b_f16(Ch, NDIM, cg * 64 + t * 16, k0, lane);
      acc[0][t] = WMMA(a0, b, acc[0][t]);
      acc[1][t] = WMMA(a1, b, acc[1][t]);
    }
  }
  #pragma unroll
  for (int s = 0; s < 2; ++s) {
    int mbase = row0 + msr + s * 16 + 8 * (lane >> 4);
    #pragma unroll
    for (int t = 0; t < 4; ++t) {
      int h = cg * 64 + t * 16 + (lane & 15);
      float d = D[h];
      #pragma unroll
      for (int r = 0; r < 8; ++r) {
        size_t idx = (size_t)(mbase + r) * HDIM + h;
        float y = acc[s][t][r] + d * x[idx];
        zb[idx] = (_Float16)(y * rsqrtf(1.0f + y * y));
      }
    }
  }
}

// ----- K4: out = z @ W^T + bias + x -----------------------------------------
// Block: 128 M-rows x 128 H.  z tile (128x128 f16 = 32KB) staged in LDS.
__global__ __launch_bounds__(256) void k4_gemm_out(
    const _Float16* __restrict__ zb, const _Float16* __restrict__ Wh,
    const float* __restrict__ bias, const float* __restrict__ x,
    float* __restrict__ out) {
  __shared__ _Float16 zs[128 * HDIM];     // 32 KB
  int tid  = threadIdx.x;
  int wave = tid >> 5, lane = tid & 31;
  int row0 = blockIdx.x * 128;
  #pragma unroll
  for (int j = 0; j < 4; ++j) {           // 1024 chunks of 16 halfs
    int chunk = tid + j * 256;
    int r = chunk >> 3, c = (chunk & 7) * 16;
    *(v16h*)(zs + r * HDIM + c) =
        *(const v16h*)(zb + (size_t)(row0 + r) * HDIM + c);
  }
  __syncthreads();

  int cg  = wave & 1;
  int msr = (wave >> 1) * 32;
  v8f acc[2][4] = {};
  #pragma unroll
  for (int k0 = 0; k0 < HDIM; k0 += 32) {
    v16h a0 = load_a_f16(zs, HDIM, msr,      k0, lane);
    v16h a1 = load_a_f16(zs, HDIM, msr + 16, k0, lane);
    #pragma unroll
    for (int t = 0; t < 4; ++t) {
      v16h b = load_b_f16(Wh, HDIM, cg * 64 + t * 16, k0, lane);
      acc[0][t] = WMMA(a0, b, acc[0][t]);
      acc[1][t] = WMMA(a1, b, acc[1][t]);
    }
  }
  #pragma unroll
  for (int s = 0; s < 2; ++s) {
    int mbase = row0 + msr + s * 16 + 8 * (lane >> 4);
    #pragma unroll
    for (int t = 0; t < 4; ++t) {
      int h = cg * 64 + t * 16 + (lane & 15);
      float bb = bias[h];
      #pragma unroll
      for (int r = 0; r < 8; ++r) {
        size_t idx = (size_t)(mbase + r) * HDIM + h;
        out[idx] = acc[s][t][r] + bb + x[idx];
      }
    }
  }
}

// ---------------------------------------------------------------------------
extern "C" void kernel_launch(void* const* d_in, const int* in_sizes, int n_in,
                              void* d_out, int out_size, void* d_ws, size_t ws_size,
                              hipStream_t stream) {
  const float* x         = (const float*)d_in[0];   // [B,L,H]
  const float* nu_log    = (const float*)d_in[1];   // [N]
  const float* gamma_log = (const float*)d_in[2];   // [N]
  const float* Bm        = (const float*)d_in[3];   // [N,H]
  const float* C         = (const float*)d_in[4];   // [H,N]
  const float* D         = (const float*)d_in[5];   // [H]
  const float* W         = (const float*)d_in[6];   // [H,H]
  const float* bias      = (const float*)d_in[7];   // [H]
  float* out = (float*)d_out;

  char* ws = (char*)d_ws;
  float*    Bu    = (float*)(ws + 0);                  // 67,108,864 B
  _Float16* hs    = (_Float16*)(ws + 67108864);        // 33,554,432 B
  _Float16* xh    = (_Float16*)(ws + 100663296);       // 16,777,216 B
  float*    ends  = (float*)(ws + 117440512);          //    262,144 B
  float*    carry = (float*)(ws + 117702656);          //    262,144 B
  _Float16* Bmh   = (_Float16*)(ws + 117964800);       //     65,536 B
  _Float16* Chh   = (_Float16*)(ws + 118030336);       //     65,536 B
  _Float16* Wh    = (_Float16*)(ws + 118095872);       //     32,768 B
  _Float16* zbuf  = (_Float16*)(ws + 0);               // alias Bu (dead after k2c)

  // f32 -> f16 operand conversion
  kcvt<<<BL * HDIM / 256, 256, 0, stream>>>(x,  xh,  BL * HDIM);
  kcvt<<<NDIM * HDIM / 256, 256, 0, stream>>>(Bm, Bmh, NDIM * HDIM);
  kcvt<<<HDIM * NDIM / 256, 256, 0, stream>>>(C,  Chh, HDIM * NDIM);
  kcvt<<<HDIM * HDIM / 256, 256, 0, stream>>>(W,  Wh,  HDIM * HDIM);

  // GEMM1: 1024 blocks of 64Mx256N (TDM-staged A tiles)
  k1_gemm_bu<<<BL / 64, 256, 0, stream>>>(xh, Bmh, gamma_log, Bu);

  // 3-phase chunked scan: 8 b x 32 chunks x 8 n-groups = 2048 waves
  k2a_ends <<<2048 / 8, 256, 0, stream>>>(Bu, nu_log, ends);
  k2b_carry<<<  64 / 8, 256, 0, stream>>>(ends, nu_log, carry);
  k2c_apply<<<2048 / 8, 256, 0, stream>>>(Bu, nu_log, carry, hs);

  // GEMM2 + D*x + tanh-approx: 512 blocks of 128Mx128H
  k3_gemm_y<<<BL / 128, 256, 0, stream>>>(hs, Chh, D, x, zbuf);

  // GEMM3 + bias + residual: 512 blocks of 128Mx128H
  k4_gemm_out<<<BL / 128, 256, 0, stream>>>(zbuf, Wh, bias, x, out);
}
